// Decoder_7730941132958
// MI455X (gfx1250) — matrix operand order repaired and checked by reading, not compiled, not run on hardware
//
#include <hip/hip_runtime.h>
#include <hip/hip_bf16.h>
#include <math.h>

typedef __attribute__((ext_vector_type(16))) _Float16 v16h;
typedef __attribute__((ext_vector_type(8)))  _Float16 v8h;
typedef __attribute__((ext_vector_type(8)))  float    v8f;
typedef __attribute__((ext_vector_type(4)))  float    f4;

#define TSTEPS 32
#define B_     64
#define S_     256
#define WV_    512
#define ENC_   512
#define DEC_   512
#define ATT_   512
#define IN_    1024   // WV + ENC (input_feed)
#define G3_    1536   // 3*DEC
#define RIN_   1536   // WV + DEC + ENC (readout input)

#if defined(__HIP_DEVICE_COMPILE__) && __has_builtin(__builtin_amdgcn_tanhf)
#define FAST_TANH(x) __builtin_amdgcn_tanhf(x)
#else
#define FAST_TANH(x) tanhf(x)
#endif

__device__ __forceinline__ float sigmoid_(float x) { return 1.0f / (1.0f + __expf(-x)); }

// ---------------------------------------------------------------------------
// WMMA fragment helpers (CDNA5 ISA 7.12.2, 16-bit layouts, wave32)
//  A 16x32: lanes 0-15 -> row=lane, K {0-7,16-23}; lanes 16-31 -> +8
//  B 32x16: lanes 0-15 -> col=lane, K {0-7,16-23}; lanes 16-31 -> +8
//  C 16x16: VGPR r -> M=r (lanes 0-15) / M=8+r (lanes 16-31), N=lane%16
// ---------------------------------------------------------------------------
__device__ __forceinline__ v16h frag_a_f16(const _Float16* p) {
  v8h lo = *(const v8h*)(p);
  v8h hi = *(const v8h*)(p + 16);
  return __builtin_shufflevector(lo, hi, 0,1,2,3,4,5,6,7,8,9,10,11,12,13,14,15);
}

// A fragment from fp32 source, converting in-register (fuses f32->f16 staging)
__device__ __forceinline__ v16h frag_a_f32(const float* p) {
  f4 a0 = *(const f4*)(p);
  f4 a1 = *(const f4*)(p + 4);
  f4 a2 = *(const f4*)(p + 16);
  f4 a3 = *(const f4*)(p + 20);
  v16h a;
#pragma unroll
  for (int i = 0; i < 4; ++i) {
    a[i]      = (_Float16)a0[i];
    a[4 + i]  = (_Float16)a1[i];
    a[8 + i]  = (_Float16)a2[i];
    a[12 + i] = (_Float16)a3[i];
  }
  return a;
}

#define WMMA(a, b, c) __builtin_amdgcn_wmma_f32_16x16x32_f16(false, a, false, b, (short)0, c, false, false)

// One contiguous-K segment of the fp32-A GEMM: invariant pointers in the hot
// loop (no per-chunk source-select branches), 4 independent WMMA chains.
__device__ __forceinline__ void gemm_seg(
    v8f& a00, v8f& a01, v8f& a10, v8f& a11,
    const float* __restrict__ src, int rl, int kw0,
    const _Float16* __restrict__ bw0, const _Float16* __restrict__ bw1,
    int mA0, int koffA) {
  if (rl <= 0) return;
  const float* ar0 = src + (size_t)mA0 * rl + koffA;
  const float* ar1 = src + (size_t)(mA0 + 16) * rl + koffA;
  const _Float16* b0p = bw0 + kw0;
  const _Float16* b1p = bw1 + kw0;
#pragma unroll 2
  for (int k = 0; k < rl; k += 32) {
    v16h a0 = frag_a_f32(ar0 + k);
    v16h a1 = frag_a_f32(ar1 + k);
    v16h b0 = frag_a_f16(b0p + k);
    v16h b1 = frag_a_f16(b1p + k);
    a00 = WMMA(a0, b0, a00);
    a01 = WMMA(a0, b1, a01);
    a10 = WMMA(a1, b0, a10);
    a11 = WMMA(a1, b1, a11);
  }
}

// ---------------------------------------------------------------------------
// 32x32-tile GEMM, A = up to 3 fp32 sources concatenated along K
// (boundaries l0, l01, multiples of 32). C[M,N] = concat(A) @ W[N,K]^T + bias.
// If gmax != null: pairwise-max (maxout POOL=2) epilogue into gmax[M, N/2].
// ---------------------------------------------------------------------------
__device__ __forceinline__ void gemm_core_af32(
    const float* __restrict__ p0, const float* __restrict__ p1,
    const float* __restrict__ p2, int l0, int l01, int K, int N,
    const _Float16* __restrict__ W, const float* __restrict__ bias,
    float* __restrict__ C, float* __restrict__ gmax, int n0, int m0) {
  const int lane  = threadIdx.x;
  const int mA0   = m0 + (lane & 15);
  const int koffA = (lane & 16) >> 1;   // 0 or 8
  const int nB0   = n0 + (lane & 15);
  const int koffB = (lane & 16) >> 1;   // 0 or 8

  const _Float16* bw0 = W + (size_t)nB0 * K + koffB;
  const _Float16* bw1 = W + (size_t)(nB0 + 16) * K + koffB;

  v8f acc00 = {}, acc01 = {}, acc10 = {}, acc11 = {};
  gemm_seg(acc00, acc01, acc10, acc11, p0, l0,       0,   bw0, bw1, mA0, koffA);
  gemm_seg(acc00, acc01, acc10, acc11, p1, l01 - l0, l0,  bw0, bw1, mA0, koffA);
  gemm_seg(acc00, acc01, acc10, acc11, p2, K - l01,  l01, bw0, bw1, mA0, koffA);

  const int nn0 = n0 + (lane & 15), nn1 = nn0 + 16;
  const int mb  = m0 + ((lane & 16) >> 1);
  const float bv0 = bias ? bias[nn0] : 0.0f;
  const float bv1 = bias ? bias[nn1] : 0.0f;
  if (!gmax) {
#pragma unroll
    for (int r = 0; r < 8; ++r) {
      C[(size_t)(mb + r) * N + nn0]      = acc00[r] + bv0;
      C[(size_t)(mb + r) * N + nn1]      = acc01[r] + bv1;
      C[(size_t)(mb + 16 + r) * N + nn0] = acc10[r] + bv0;
      C[(size_t)(mb + 16 + r) * N + nn1] = acc11[r] + bv1;
    }
  } else {
    const int NH = N >> 1;
#pragma unroll
    for (int r = 0; r < 8; ++r) {
      float v00 = acc00[r] + bv0, v01 = acc01[r] + bv1;
      float v10 = acc10[r] + bv0, v11 = acc11[r] + bv1;
      float w00 = __shfl_xor(v00, 1, 32), w01 = __shfl_xor(v01, 1, 32);
      float w10 = __shfl_xor(v10, 1, 32), w11 = __shfl_xor(v11, 1, 32);
      if ((lane & 1) == 0) {   // even lane owns the pair (n, n+1)
        gmax[(size_t)(mb + r) * NH + (nn0 >> 1)]      = fmaxf(v00, w00);
        gmax[(size_t)(mb + r) * NH + (nn1 >> 1)]      = fmaxf(v01, w01);
        gmax[(size_t)(mb + 16 + r) * NH + (nn0 >> 1)] = fmaxf(v10, w10);
        gmax[(size_t)(mb + 16 + r) * NH + (nn1 >> 1)] = fmaxf(v11, w11);
      }
    }
  }
}

__global__ void k_gemm_af32(const float* __restrict__ p0, const float* __restrict__ p1,
                            const float* __restrict__ p2, int l0, int l01, int K, int N,
                            const _Float16* __restrict__ W, const float* __restrict__ bias,
                            float* __restrict__ C, float* __restrict__ gmax) {
  gemm_core_af32(p0, p1, p2, l0, l01, K, N, W, bias, C, gmax,
                 blockIdx.x * 32, blockIdx.y * 32);
}

// Both GRU gemms in ONE launch: z=0 -> G1 = [emb|ctx] @ W_ih^T + b_ih (K=1024)
//                               z=1 -> G2 = h @ W_hh^T + b_hh        (K=512)
__global__ void k_gru_gemms(const float* __restrict__ emb_t, const float* __restrict__ cc,
                            const float* __restrict__ h,
                            const _Float16* __restrict__ Wih, const _Float16* __restrict__ Whh,
                            const float* __restrict__ bih, const float* __restrict__ bhh,
                            float* __restrict__ G1, float* __restrict__ G2) {
  if (blockIdx.z == 0)
    gemm_core_af32(emb_t, cc, cc, WV_, IN_, IN_, G3_, Wih, bih, G1, nullptr,
                   blockIdx.x * 32, blockIdx.y * 32);
  else
    gemm_core_af32(h, h, h, DEC_, DEC_, DEC_, G3_, Whh, bhh, G2, nullptr,
                   blockIdx.x * 32, blockIdx.y * 32);
}

// f16-A GEMM (32x32 tile) for the one-time attention precompute
__global__ void k_gemm_f16(const _Float16* __restrict__ A, const _Float16* __restrict__ W,
                           const float* __restrict__ bias, float* __restrict__ C,
                           int K, int N) {
  const int lane  = threadIdx.x;
  const int n0 = blockIdx.x * 32, m0 = blockIdx.y * 32;
  const int mA0   = m0 + (lane & 15);
  const int koffA = (lane & 16) >> 1;
  const int nB0   = n0 + (lane & 15);
  const int koffB = (lane & 16) >> 1;
  const _Float16* ar0 = A + (size_t)mA0 * K + koffA;
  const _Float16* ar1 = A + (size_t)(mA0 + 16) * K + koffA;
  const _Float16* bw0 = W + (size_t)nB0 * K + koffB;
  const _Float16* bw1 = W + (size_t)(nB0 + 16) * K + koffB;
  v8f acc00 = {}, acc01 = {}, acc10 = {}, acc11 = {};
#pragma unroll 2
  for (int k0 = 0; k0 < K; k0 += 32) {
    v16h a0 = frag_a_f16(ar0 + k0);
    v16h a1 = frag_a_f16(ar1 + k0);
    v16h b0 = frag_a_f16(bw0 + k0);
    v16h b1 = frag_a_f16(bw1 + k0);
    acc00 = WMMA(a0, b0, acc00);
    acc01 = WMMA(a0, b1, acc01);
    acc10 = WMMA(a1, b0, acc10);
    acc11 = WMMA(a1, b1, acc11);
  }
  const int nn0 = n0 + (lane & 15), nn1 = nn0 + 16;
  const int mb  = m0 + ((lane & 16) >> 1);
  const float bv0 = bias[nn0], bv1 = bias[nn1];
#pragma unroll
  for (int r = 0; r < 8; ++r) {
    C[(size_t)(mb + r) * N + nn0]      = acc00[r] + bv0;
    C[(size_t)(mb + r) * N + nn1]      = acc01[r] + bv1;
    C[(size_t)(mb + 16 + r) * N + nn0] = acc10[r] + bv0;
    C[(size_t)(mb + 16 + r) * N + nn1] = acc11[r] + bv1;
  }
}

// ---------------------------------------------------------------------------
// One-time staging kernels
// ---------------------------------------------------------------------------
__global__ void k_cvt_all(const float* __restrict__ Wih, const float* __restrict__ Whh,
                          const float* __restrict__ Wpre, const float* __restrict__ Wq,
                          const float* __restrict__ Wread,
                          _Float16* __restrict__ o_ih, _Float16* __restrict__ o_hh,
                          _Float16* __restrict__ o_pre, _Float16* __restrict__ o_q,
                          _Float16* __restrict__ o_read) {
  int i = blockIdx.x * blockDim.x + threadIdx.x;
  const int n0 = G3_ * IN_;
  const int n1 = n0 + G3_ * DEC_;
  const int n2 = n1 + ATT_ * ENC_;
  const int n3 = n2 + ATT_ * DEC_;
  const int n4 = n3 + DEC_ * RIN_;
  if (i < n0)      o_ih[i]        = (_Float16)Wih[i];
  else if (i < n1) o_hh[i - n0]   = (_Float16)Whh[i - n0];
  else if (i < n2) o_pre[i - n1]  = (_Float16)Wpre[i - n1];
  else if (i < n3) o_q[i - n2]    = (_Float16)Wq[i - n2];
  else if (i < n4) o_read[i - n3] = (_Float16)Wread[i - n3];
}

// context [S,B,E] f32 -> ctxh [B*S, E] f16 (batch-major rows)
__global__ void k_ctx_to_f16(const float* __restrict__ ctx, _Float16* __restrict__ d) {
  int i = blockIdx.x * blockDim.x + threadIdx.x;
  if (i >= S_ * B_ * ENC_) return;
  int s   = i / (B_ * ENC_);
  int rem = i - s * (B_ * ENC_);
  int b   = rem / ENC_;
  int e   = rem - b * ENC_;
  d[((size_t)b * S_ + s) * ENC_ + e] = (_Float16)ctx[i];
}

__global__ void k_embed(const int* __restrict__ ids, const float* __restrict__ table,
                        float* __restrict__ emb) {
  int row = blockIdx.x;
  int id = ids[row];
  const float* src = table + (size_t)id * WV_;
  float* dst = emb + (size_t)row * WV_;
  for (int i = threadIdx.x; i < WV_; i += blockDim.x) dst[i] = src[i];
}

__global__ void k_init(const float* __restrict__ hidden, const float* __restrict__ init_att,
                       float* __restrict__ h, float* __restrict__ cc) {
  int i = blockIdx.x * blockDim.x + threadIdx.x;
  if (i >= B_ * DEC_) return;
  h[i]  = hidden[i];
  cc[i] = init_att[i];
}

// ---------------------------------------------------------------------------
// Per-step fused kernels
// ---------------------------------------------------------------------------
__global__ void k_gru_gates(const float* __restrict__ G1, const float* __restrict__ G2,
                            float* __restrict__ h) {
  int i = blockIdx.x * blockDim.x + threadIdx.x;
  if (i >= B_ * DEC_) return;
  int b = i >> 9, n = i & (DEC_ - 1);
  const float* g1 = G1 + (size_t)b * G3_;
  const float* g2 = G2 + (size_t)b * G3_;
  float r  = sigmoid_(g1[n] + g2[n]);
  float z  = sigmoid_(g1[DEC_ + n] + g2[DEC_ + n]);
  float nn = FAST_TANH(g1[2 * DEC_ + n] + r * g2[2 * DEC_ + n]);
  h[i] = (1.0f - z) * nn + z * h[i];
}

// energy[b,s] = dot(v_att, tanh(pre[b,s,:] + target[b,:])), masked.
// 8 waves/block, one (b,s) row per wave, float4 loads, hw v_tanh_f32.
__global__ void k_energy(const float* __restrict__ pre, const float* __restrict__ target,
                         const float* __restrict__ v_att, const float* __restrict__ mask,
                         float* __restrict__ energy) {
  int lane = threadIdx.x & 31, wv = threadIdx.x >> 5;
  int row = blockIdx.x * 8 + wv;
  int b = row >> 8, s = row & (S_ - 1);
  const float* p = pre + (size_t)row * ATT_;
  const float* t = target + (size_t)b * ATT_;
  float acc = 0.0f;
  for (int a = lane * 4; a < ATT_; a += 128) {
    f4 pv = *(const f4*)(p + a);
    f4 tv = *(const f4*)(t + a);
    f4 vv = *(const f4*)(v_att + a);
#pragma unroll
    for (int j = 0; j < 4; ++j) acc += vv[j] * FAST_TANH(pv[j] + tv[j]);
  }
  for (int off = 16; off; off >>= 1) acc += __shfl_down(acc, off, 32);
  if (lane == 0) {
    float m = mask[b * S_ + s];
    energy[row] = acc * (1.0f - m) + m * (-1e6f);
  }
}

// Per-b: softmax over S, new_ctx = score @ ctx, AND copy gate (fused).
__global__ void k_attn(const float* __restrict__ energy, const float* __restrict__ context,
                       const float* __restrict__ h,
                       const float* __restrict__ Wc, const float* __restrict__ bc,
                       float* __restrict__ score_out, float* __restrict__ cur_ctx,
                       float* __restrict__ copy_out) {
  __shared__ float sc[S_];
  __shared__ float red[8];
  const int b = blockIdx.x, tid = threadIdx.x;
  const int lane = tid & 31, wv = tid >> 5;

  float e = energy[b * S_ + tid];
  float m = e;
  for (int off = 16; off; off >>= 1) m = fmaxf(m, __shfl_xor(m, off, 32));
  if (lane == 0) red[wv] = m;
  __syncthreads();
  if (tid == 0) {
    float mm = red[0];
#pragma unroll
    for (int i = 1; i < 8; ++i) mm = fmaxf(mm, red[i]);
    red[0] = mm;
  }
  __syncthreads();
  const float mx = red[0];
  float ex = __expf(e - mx);
  float sm = ex;
  for (int off = 16; off; off >>= 1) sm += __shfl_xor(sm, off, 32);
  __syncthreads();
  if (lane == 0) red[wv] = sm;
  __syncthreads();
  if (tid == 0) {
    float ss = 0.0f;
#pragma unroll
    for (int i = 0; i < 8; ++i) ss += red[i];
    red[0] = ss;
  }
  __syncthreads();
  const float p = ex / red[0];
  sc[tid] = p;
  score_out[b * S_ + tid] = p;
  __syncthreads();

  // new_ctx + partial copy-gate dot
  float cpart = 0.0f;
  for (int e2 = tid; e2 < ENC_; e2 += 256) {
    float acc = 0.0f;
    for (int s2 = 0; s2 < S_; ++s2)
      acc += sc[s2] * context[((size_t)s2 * B_ + b) * ENC_ + e2];
    cur_ctx[b * ENC_ + e2] = acc;
    cpart += acc * Wc[DEC_ + e2];
  }
  for (int i = tid; i < DEC_; i += 256) cpart += h[b * DEC_ + i] * Wc[i];
  for (int off = 16; off; off >>= 1) cpart += __shfl_down(cpart, off, 32);
  __syncthreads();
  if (lane == 0) red[wv] = cpart;
  __syncthreads();
  if (tid == 0) {
    float ss = 0.0f;
#pragma unroll
    for (int i = 0; i < 8; ++i) ss += red[i];
    copy_out[b] = sigmoid_(ss + bc[0]);
  }
}

// ---------------------------------------------------------------------------
extern "C" void kernel_launch(void* const* d_in, const int* in_sizes, int n_in,
                              void* d_out, int out_size, void* d_ws, size_t ws_size,
                              hipStream_t stream) {
  (void)in_sizes; (void)n_in; (void)out_size; (void)ws_size;
  const int*   ids      = (const int*)d_in[0];
  const float* hidden   = (const float*)d_in[1];
  const float* context  = (const float*)d_in[2];
  const float* mask     = (const float*)d_in[3];
  const float* init_att = (const float*)d_in[4];
  const float* table    = (const float*)d_in[5];
  const float* W_ih     = (const float*)d_in[6];
  const float* W_hh     = (const float*)d_in[7];
  const float* b_ih     = (const float*)d_in[8];
  const float* b_hh     = (const float*)d_in[9];
  const float* W_pre    = (const float*)d_in[10];
  const float* b_pre    = (const float*)d_in[11];
  const float* W_q      = (const float*)d_in[12];
  const float* v_att    = (const float*)d_in[13];
  const float* W_copy   = (const float*)d_in[14];
  const float* b_copy   = (const float*)d_in[15];
  const float* W_read   = (const float*)d_in[16];
  const float* b_read   = (const float*)d_in[17];

  float* out = (float*)d_out;
  const size_t g_off     = 0;                                   // [T,B,DEC/2]
  const size_t c_off     = g_off + (size_t)TSTEPS * B_ * (DEC_ / 2);
  const size_t cp_off    = c_off + (size_t)TSTEPS * B_ * S_;    // [T,B,1]
  const size_t hid_off   = cp_off + (size_t)TSTEPS * B_;        // [1,B,DEC]
  const size_t clast_off = hid_off + (size_t)B_ * DEC_;         // [B,S]
  const size_t cc_off    = clast_off + (size_t)B_ * S_;         // [B,ENC]

  char* wp = (char*)d_ws;
  auto alloc = [&](size_t bytes) -> void* {
    void* r = (void*)wp;
    wp += (bytes + 255) & ~(size_t)255;
    return r;
  };
  _Float16* W_ihh   = (_Float16*)alloc(sizeof(_Float16) * G3_ * IN_);
  _Float16* W_hhh   = (_Float16*)alloc(sizeof(_Float16) * G3_ * DEC_);
  _Float16* W_preh  = (_Float16*)alloc(sizeof(_Float16) * ATT_ * ENC_);
  _Float16* W_qh    = (_Float16*)alloc(sizeof(_Float16) * ATT_ * DEC_);
  _Float16* W_readh = (_Float16*)alloc(sizeof(_Float16) * DEC_ * RIN_);
  _Float16* ctxh    = (_Float16*)alloc(sizeof(_Float16) * (size_t)B_ * S_ * ENC_);
  float*    pre     = (float*)alloc(sizeof(float) * (size_t)B_ * S_ * ATT_);
  float*    embf    = (float*)alloc(sizeof(float) * (size_t)TSTEPS * B_ * WV_);
  float*    G1      = (float*)alloc(sizeof(float) * B_ * G3_);
  float*    G2      = (float*)alloc(sizeof(float) * B_ * G3_);
  float*    hbuf    = (float*)alloc(sizeof(float) * B_ * DEC_);
  float*    ccbuf   = (float*)alloc(sizeof(float) * B_ * ENC_);
  float*    target  = (float*)alloc(sizeof(float) * B_ * ATT_);
  float*    energy  = (float*)alloc(sizeof(float) * B_ * S_);

  // --- one-time setup (5 launches) ---
  const int NCVT = G3_ * IN_ + G3_ * DEC_ + ATT_ * ENC_ + ATT_ * DEC_ + DEC_ * RIN_;
  k_cvt_all<<<(NCVT + 255) / 256, 256, 0, stream>>>(
      W_ih, W_hh, W_pre, W_q, W_read, W_ihh, W_hhh, W_preh, W_qh, W_readh);
  k_ctx_to_f16<<<(B_ * S_ * ENC_ + 255) / 256, 256, 0, stream>>>(context, ctxh);
  k_embed<<<TSTEPS * B_, 128, 0, stream>>>(ids, table, embf);
  k_init<<<(B_ * DEC_ + 255) / 256, 256, 0, stream>>>(hidden, init_att, hbuf, ccbuf);
  // precompute[b,s,:] = ctx_t @ W_pre^T + b_pre  (M=16384, N=512, K=512)
  k_gemm_f16<<<dim3(ATT_ / 32, (B_ * S_) / 32), 32, 0, stream>>>(
      ctxh, W_preh, b_pre, pre, ENC_, ATT_);

  // --- 32 dependent steps, 6 launches each ---
  for (int t = 0; t < TSTEPS; ++t) {
    const float* emb_t = embf + (size_t)t * B_ * WV_;
    k_gru_gemms<<<dim3(G3_ / 32, B_ / 32, 2), 32, 0, stream>>>(
        emb_t, ccbuf, hbuf, W_ihh, W_hhh, b_ih, b_hh, G1, G2);
    k_gru_gates<<<(B_ * DEC_ + 255) / 256, 256, 0, stream>>>(G1, G2, hbuf);
    k_gemm_af32<<<dim3(ATT_ / 32, B_ / 32), 32, 0, stream>>>(
        hbuf, hbuf, hbuf, DEC_, DEC_, DEC_, ATT_, W_qh, nullptr, target, nullptr);
    k_energy<<<(B_ * S_) / 8, 256, 0, stream>>>(pre, target, v_att, mask, energy);
    k_attn<<<B_, S_, 0, stream>>>(energy, context, hbuf, W_copy, b_copy,
                                  out + c_off + (size_t)t * B_ * S_, ccbuf,
                                  out + cp_off + (size_t)t * B_);
    // readout GEMM with fused maxout epilogue, A = [emb_t | h | new_ctx] (fp32)
    k_gemm_af32<<<dim3(DEC_ / 32, B_ / 32), 32, 0, stream>>>(
        emb_t, hbuf, ccbuf, WV_, WV_ + DEC_, RIN_, DEC_, W_readh, b_read,
        nullptr, out + g_off + (size_t)t * B_ * (DEC_ / 2));
  }

  hipMemcpyAsync(out + hid_off, hbuf, sizeof(float) * B_ * DEC_,
                 hipMemcpyDeviceToDevice, stream);
  hipMemcpyAsync(out + clast_off, out + c_off + (size_t)(TSTEPS - 1) * B_ * S_,
                 sizeof(float) * B_ * S_, hipMemcpyDeviceToDevice, stream);
  hipMemcpyAsync(out + cc_off, ccbuf, sizeof(float) * B_ * ENC_,
                 hipMemcpyDeviceToDevice, stream);
}
